// WakeCorrectorGNN_14018773254834
// MI455X (gfx1250) — compile-verified, matrix operand order checked by reading, not run on hardware
//
#include <hip/hip_runtime.h>
#include <math.h>

// ---------------- problem constants ----------------
#define B_   2
#define K_   2
#define N_   4096
#define T_   8
#define C_   3
#define H_   192
#define S_   1638            // int(0.4*4096)
#define KNN_ 16
#define E_   (S_*KNN_)       // 26208
#define G_   (B_*K_)         // 4 graphs
#define F0_  24              // 2*10+4
#define FR_  388             // 2*192+4
#define KP0_ 32              // F0 padded to mult of 32
#define KPR_ 416             // FR padded to mult of 32
#define TILE_E 64
#define SHARP_ 5.0f

typedef __attribute__((ext_vector_type(16))) _Float16 v16h;
typedef __attribute__((ext_vector_type(8)))  float    v8f;

union FragH { v16h h; unsigned int u[8]; };

// CDNA5 16-bit A/B fragment K mapping (ISA 7.12.2): lane half selects K groups.
__device__ __forceinline__ int kmap_(int v, int hf) {
    return (v < 4) ? (2*v + 8*hf) : (16 + 2*(v-4) + 8*hf);
}
__device__ __forceinline__ void load_frag(FragH& f, const _Float16* base, int hf) {
#pragma unroll
    for (int v = 0; v < 8; ++v)
        f.u[v] = *(const unsigned int*)(base + kmap_(v, hf));
}
__device__ __forceinline__ float gelu_(float x) {
    return 0.5f * x * (1.0f + erff(x * 0.70710678118654752f));
}

// ---------------- stage 1: variance over time ----------------
__global__ void varpp_kernel(const float* __restrict__ vin, float* __restrict__ varpp) {
    int idx = blockIdx.x * blockDim.x + threadIdx.x;
    if (idx >= B_ * N_) return;
    int b = idx / N_, n = idx % N_;
    float acc = 0.f;
#pragma unroll
    for (int c = 0; c < C_; ++c) {
        float s = 0.f, s2 = 0.f;
#pragma unroll
        for (int t = 0; t < T_; ++t) {
            float x = vin[(((size_t)b*T_ + t)*N_ + n)*C_ + c];
            s += x; s2 += x*x;
        }
        float mu = s / (float)T_;
        acc += (s2 - (float)T_*mu*mu) / (float)(T_ - 1);
    }
    varpp[idx] = acc;
}

// ---------------- stage 2: per-batch mean/std (ddof=1) ----------------
__global__ void stats_kernel(const float* __restrict__ varpp, float* __restrict__ stats) {
    __shared__ float r1[256], r2[256];
    int b = blockIdx.x, tid = threadIdx.x;
    float s = 0.f, s2 = 0.f;
    for (int n = tid; n < N_; n += 256) {
        float x = varpp[b*N_ + n];
        s += x; s2 += x*x;
    }
    r1[tid] = s; r2[tid] = s2;
    __syncthreads();
    for (int off = 128; off > 0; off >>= 1) {
        if (tid < off) { r1[tid] += r1[tid+off]; r2[tid] += r2[tid+off]; }
        __syncthreads();
    }
    if (tid == 0) {
        float mean = r1[0] / (float)N_;
        float var  = (r2[0] - (float)N_*mean*mean) / (float)(N_ - 1);
        stats[b*2+0] = mean;
        stats[b*2+1] = sqrtf(fmaxf(var, 0.f));
    }
}

// ---------------- stage 3: soft mask ----------------
__global__ void softmask_kernel(const float* __restrict__ varpp, const float* __restrict__ stats,
                                const float* __restrict__ am, float* __restrict__ smask) {
    int idx = blockIdx.x * blockDim.x + threadIdx.x;
    if (idx >= B_ * N_) return;
    int b = idx / N_;
    float z = (varpp[idx] - stats[b*2+0]) / (stats[b*2+1] + 1e-8f);
    float sg = 1.f / (1.f + expf(-SHARP_ * z));
    smask[idx] = sg * (1.f - am[idx]);
}

// ---------------- stage 4: top-S via bitonic sort (descending) ----------------
__global__ __launch_bounds__(1024) void topk_sort_kernel(const float* __restrict__ varpp,
                                                         int* __restrict__ topidx) {
    __shared__ float key[N_];
    __shared__ int   ids[N_];
    int b = blockIdx.x;
    for (int i = threadIdx.x; i < N_; i += blockDim.x) { key[i] = varpp[b*N_ + i]; ids[i] = i; }
    __syncthreads();
    for (int ks = 2; ks <= N_; ks <<= 1) {
        for (int j = ks >> 1; j > 0; j >>= 1) {
            for (int i = threadIdx.x; i < N_; i += blockDim.x) {
                int l = i ^ j;
                if (l > i) {
                    bool up = ((i & ks) == 0);           // descending overall
                    float ki = key[i], kl = key[l];
                    bool sw = up ? (ki < kl) : (ki > kl);
                    if (sw) {
                        key[i] = kl; key[l] = ki;
                        int t = ids[i]; ids[i] = ids[l]; ids[l] = t;
                    }
                }
            }
            __syncthreads();
        }
    }
    for (int s = threadIdx.x; s < S_; s += blockDim.x) topidx[b*S_ + s] = ids[s];
}

// ---------------- stage 5: gather selected nodes, build x0 ----------------
__global__ void gather_kernel(const float* __restrict__ u_base, const float* __restrict__ pos,
                              const float* __restrict__ vin, const float* __restrict__ varpp,
                              const float* __restrict__ smask, const int* __restrict__ topidx,
                              float* __restrict__ possel, float* __restrict__ masksel,
                              float* __restrict__ x0) {
    int idx = blockIdx.x * blockDim.x + threadIdx.x;
    if (idx >= B_ * S_) return;
    int b = idx / S_, s = idx % S_;
    int n = topidx[b*S_ + s];
    float p0 = pos[((size_t)b*N_ + n)*3 + 0];
    float p1 = pos[((size_t)b*N_ + n)*3 + 1];
    float p2 = pos[((size_t)b*N_ + n)*3 + 2];
    possel[(size_t)idx*3+0] = p0; possel[(size_t)idx*3+1] = p1; possel[(size_t)idx*3+2] = p2;
    masksel[idx] = smask[b*N_ + n];
    float vl0 = vin[(((size_t)b*T_ + (T_-1))*N_ + n)*C_ + 0];
    float vl1 = vin[(((size_t)b*T_ + (T_-1))*N_ + n)*C_ + 1];
    float vl2 = vin[(((size_t)b*T_ + (T_-1))*N_ + n)*C_ + 2];
    float vp  = varpp[b*N_ + n];
#pragma unroll
    for (int k = 0; k < K_; ++k) {
        float* xp = x0 + (((size_t)(b*K_ + k))*S_ + s)*10;
        xp[0] = u_base[(((size_t)(b*K_ + k))*N_ + n)*C_ + 0];
        xp[1] = u_base[(((size_t)(b*K_ + k))*N_ + n)*C_ + 1];
        xp[2] = u_base[(((size_t)(b*K_ + k))*N_ + n)*C_ + 2];
        xp[3] = vl0; xp[4] = vl1; xp[5] = vl2;
        xp[6] = p0;  xp[7] = p1;  xp[8] = p2;
        xp[9] = vp;
    }
}

// ---------------- stage 6: KNN + edge attributes (LDS-staged positions) ----------------
__global__ __launch_bounds__(256)
void knn_kernel(const float* __restrict__ possel, int* __restrict__ knnb,
                float* __restrict__ ea, _Float16* __restrict__ ea16) {
    __shared__ float spx[S_], spy[S_], spz[S_];
    int b = blockIdx.y;
    const float* pb = possel + (size_t)b*S_*3;
    for (int t = threadIdx.x; t < S_; t += blockDim.x) {
        spx[t] = pb[t*3+0]; spy[t] = pb[t*3+1]; spz[t] = pb[t*3+2];
    }
    __syncthreads();
    int s = blockIdx.x * blockDim.x + threadIdx.x;
    if (s >= S_) return;
    float px = spx[s], py = spy[s], pz = spz[s];
    float bd[KNN_]; int bi[KNN_];
#pragma unroll
    for (int k = 0; k < KNN_; ++k) { bd[k] = 3.4e38f; bi[k] = 0; }
    for (int t = 0; t < S_; ++t) {
        if (t == s) continue;
        float dx = spx[t] - px, dy = spy[t] - py, dz = spz[t] - pz;
        float d2 = dx*dx + dy*dy + dz*dz;
        if (d2 < bd[KNN_-1]) {
            int p = KNN_-1;
            while (p > 0 && bd[p-1] > d2) { bd[p] = bd[p-1]; bi[p] = bi[p-1]; --p; }
            bd[p] = d2; bi[p] = t;
        }
    }
#pragma unroll
    for (int k = 0; k < KNN_; ++k) {
        int i = bi[k];
        knnb[((size_t)b*S_ + s)*KNN_ + k] = i;
        int e = s*KNN_ + k;
        float rx = spx[i] - px, ry = spy[i] - py, rz = spz[i] - pz;
        float dd = sqrtf(rx*rx + ry*ry + rz*rz);
        float* ep = ea + ((size_t)b*E_ + e)*4;
        ep[0] = rx; ep[1] = ry; ep[2] = rz; ep[3] = dd;
        _Float16* q = ea16 + ((size_t)b*E_ + e)*32;   // padded, async-copy friendly
        q[0] = (_Float16)rx; q[1] = (_Float16)ry; q[2] = (_Float16)rz; q[3] = (_Float16)dd;
#pragma unroll
        for (int z = 4; z < 32; ++z) q[z] = (_Float16)0.f;
    }
}

// ---------------- weight prep ----------------
// plain transpose+pad: src[l][kk][n] (Kdim x H) -> dst[l][n][kk] (H x Kp)
__global__ void prep_w_kernel(const float* __restrict__ src, _Float16* __restrict__ dst,
                              int Kdim, int Kp, int layers, int src_lstride, int dst_lstride) {
    int total = layers * H_ * Kp;
    for (int i = blockIdx.x * blockDim.x + threadIdx.x; i < total; i += gridDim.x * blockDim.x) {
        int l = i / (H_ * Kp);
        int rem = i % (H_ * Kp);
        int n = rem / Kp, kk = rem % Kp;
        float v = (kk < Kdim) ? src[(size_t)l*src_lstride + (size_t)kk*H_ + n] : 0.f;
        dst[(size_t)l*dst_lstride + (size_t)n*Kp + kk] = (_Float16)v;
    }
}

// folded W1: message = [hi | hj-hi | ea] @ W1  ==  [hi | hj | ea] @ W1fold with
//   W1fold[0:D)   = W1[0:D) - W1[D:2D)
//   W1fold[D:2D+4)= W1[D:2D+4)
__global__ void prep_w1fold_kernel(const float* __restrict__ src, _Float16* __restrict__ dst,
                                   int Din, int Kp, int layers, int src_lstride, int dst_lstride) {
    int total = layers * H_ * Kp;
    for (int i = blockIdx.x * blockDim.x + threadIdx.x; i < total; i += gridDim.x * blockDim.x) {
        int l = i / (H_ * Kp);
        int rem = i % (H_ * Kp);
        int n = rem / Kp, kk = rem % Kp;
        const float* sl = src + (size_t)l*src_lstride;
        float v = 0.f;
        if (kk < Din)            v = sl[(size_t)kk*H_ + n] - sl[(size_t)(Din+kk)*H_ + n];
        else if (kk < 2*Din + 4) v = sl[(size_t)kk*H_ + n];
        dst[(size_t)l*dst_lstride + (size_t)n*Kp + kk] = (_Float16)v;
    }
}

// ---------------- WMMA fused edge MLP + scatter-add ----------------
__global__ __launch_bounds__(256)
void conv_edge_mlp(const float* __restrict__ hin, const _Float16* __restrict__ h16,
                   int Din,
                   const int* __restrict__ knn_all, const float* __restrict__ ea_all,
                   const _Float16* __restrict__ ea16_all,
                   const _Float16* __restrict__ w1T, int Kp,
                   const float* __restrict__ b1,
                   const _Float16* __restrict__ w2T,
                   const float* __restrict__ b2,
                   float* __restrict__ agg) {
    extern __shared__ char smem[];
    const int maxKp = (Kp > H_) ? Kp : H_;
    _Float16* Ab = (_Float16*)smem;      // TILE_E x Kp message tile (f16)
    _Float16* T1 = (_Float16*)smem;      // reused: TILE_E x H intermediate (f16)
    int* sidx = (int*)(smem + (size_t)TILE_E * maxKp * 2);

    const int g = blockIdx.y;
    const int b = g / K_;
    const int*   knnb = knn_all + (size_t)b * S_ * KNN_;
    float* aggg = agg + (size_t)g * S_ * H_;

    const int e0 = blockIdx.x * TILE_E;
    for (int r = threadIdx.x; r < TILE_E; r += blockDim.x) {
        int e = e0 + r;
        sidx[r] = (e < E_) ? knnb[e] : -1;  // edge e = s*KNN+k -> neighbor i
    }
    __syncthreads();

    if (Din == H_) {
        // Async gather: row = [h_i(192 f16) | h_j(192 f16) | ea16(32 f16)] = 52 x 16B chunks.
        // Pure copies (subtraction folded into W1) -> use CDNA5 async global->LDS DMA.
        const _Float16* h16g = h16 + (size_t)g * S_ * H_;
        const char* hb = (const char*)h16g;                       // 384 B per node row
        const char* eb = (const char*)(ea16_all + (size_t)b * E_ * 32); // 64 B per edge
        unsigned abase = (unsigned)(uintptr_t)Ab;
        for (int p = threadIdx.x; p < TILE_E * 52; p += blockDim.x) {
            int r = p & (TILE_E - 1);
            int c = p >> 6;                    // chunk 0..51
            int e = e0 + r;
            if (e < E_) {
                unsigned laddr = abase + (unsigned)r * 832u + (unsigned)c * 16u;
                const char* gsrc;
                if (c < 24)      gsrc = hb + (size_t)sidx[r] * 384 + (size_t)c * 16;
                else if (c < 48) gsrc = hb + (size_t)(e >> 4) * 384 + (size_t)(c - 24) * 16;
                else             gsrc = eb + (size_t)e * 64 + (size_t)(c - 48) * 16;
                asm volatile("global_load_async_to_lds_b128 %0, %1, off"
                             :: "v"(laddr), "v"(gsrc) : "memory");
            }
        }
        asm volatile("s_wait_asynccnt 0" ::: "memory");
    } else {
        // Layer 0 scalar staging (Din=10, Kp=32): [hi | hj | ea | pad] -> f16
        const float* hing = hin + (size_t)g * S_ * Din;
        const float* eab  = ea_all + (size_t)b * E_ * 4;
        for (int p = threadIdx.x; p < TILE_E * KP0_; p += blockDim.x) {
            int r = p >> 5, c = p & 31;
            int e = e0 + r;
            float v = 0.f;
            if (e < E_) {
                int s = e >> 4;
                int i = sidx[r];
                if (c < Din)            v = hing[(size_t)i*Din + c];
                else if (c < 2*Din)     v = hing[(size_t)s*Din + (c - Din)];
                else if (c < 2*Din + 4) v = eab[(size_t)e*4 + (c - 2*Din)];
            }
            Ab[(size_t)r*KP0_ + c] = (_Float16)v;
        }
    }
    __syncthreads();

    const int wave = threadIdx.x >> 5, lane = threadIdx.x & 31;
    const int hf = lane >> 4, lrow = lane & 15;
    const int rb = wave >> 1, ch = wave & 1;   // 4 row-blocks x 2 col-halves

    v8f vz = {};
    v8f acc[6];
#pragma unroll
    for (int cb = 0; cb < 6; ++cb) acc[cb] = vz;

    // GEMM1: (TILE_E x Kp) @ w1T^T -> (TILE_E x 192)
    const _Float16* Arow = Ab + (size_t)(rb*16 + lrow) * Kp;
    for (int kc = 0; kc < Kp; kc += 32) {
        FragH a; load_frag(a, Arow + kc, hf);
#pragma unroll
        for (int cb = 0; cb < 6; ++cb) {
            int cbg = ch*6 + cb;
            FragH bf; load_frag(bf, w1T + (size_t)(cbg*16 + lrow)*Kp + kc, hf);
            acc[cb] = __builtin_amdgcn_wmma_f32_16x16x32_f16(
                false, a.h, false, bf.h, (short)0, acc[cb], false, false);
        }
    }
    __syncthreads();
    // bias + exact GELU -> T1 (f16), reusing the A-tile LDS region
#pragma unroll
    for (int cb = 0; cb < 6; ++cb) {
        int col = (ch*6 + cb)*16 + lrow;
        float bb = b1[col];
#pragma unroll
        for (int r = 0; r < 8; ++r) {
            int m = r + 8*hf;
            T1[(size_t)(rb*16 + m)*H_ + col] = (_Float16)gelu_(acc[cb][r] + bb);
        }
    }
    __syncthreads();

    // GEMM2: (TILE_E x 192) @ w2T^T -> (TILE_E x 192)
#pragma unroll
    for (int cb = 0; cb < 6; ++cb) acc[cb] = vz;
    const _Float16* A2 = T1 + (size_t)(rb*16 + lrow) * H_;
    for (int kc = 0; kc < H_; kc += 32) {
        FragH a; load_frag(a, A2 + kc, hf);
#pragma unroll
        for (int cb = 0; cb < 6; ++cb) {
            int cbg = ch*6 + cb;
            FragH bf; load_frag(bf, w2T + (size_t)(cbg*16 + lrow)*H_ + kc, hf);
            acc[cb] = __builtin_amdgcn_wmma_f32_16x16x32_f16(
                false, a.h, false, bf.h, (short)0, acc[cb], false, false);
        }
    }
    // bias + scatter-add to aggregate nodes (by neighbor index i)
#pragma unroll
    for (int cb = 0; cb < 6; ++cb) {
        int col = (ch*6 + cb)*16 + lrow;
        float bb = b2[col];
#pragma unroll
        for (int r = 0; r < 8; ++r) {
            int m = r + 8*hf;
            int node = sidx[rb*16 + m];
            if (node >= 0)
                atomicAdd(&aggg[(size_t)node*H_ + col], acc[cb][r] + bb);
        }
    }
}

// ---------------- LayerNorm (+residual), emits f32 and/or f16 ----------------
// mode 0: outf = ln(inp), outh = f16(outf)
// mode 1: outf = ln(inp) + res, outh = f16(outf)
// mode 2: outh = f16(ln(inp)) only
__global__ __launch_bounds__(192)
void ln_kernel(const float* __restrict__ inp, const float* __restrict__ res,
               const float* __restrict__ g, const float* __restrict__ be,
               float* __restrict__ outf, _Float16* __restrict__ outh, int mode) {
    __shared__ float r1[H_], r2[H_];
    __shared__ float mu_s, rstd_s;
    int node = blockIdx.x, c = threadIdx.x;
    float x = inp[(size_t)node*H_ + c];
    r1[c] = x; r2[c] = x*x;
    __syncthreads();
    for (int off = 96; off >= 3; off >>= 1) {
        if (c < off) { r1[c] += r1[c+off]; r2[c] += r2[c+off]; }
        __syncthreads();
    }
    if (c == 0) {
        float s1 = r1[0]+r1[1]+r1[2], s2 = r2[0]+r2[1]+r2[2];
        float mu = s1 / (float)H_;
        float var = s2 / (float)H_ - mu*mu;
        mu_s = mu; rstd_s = rsqrtf(fmaxf(var, 0.f) + 1e-5f);
    }
    __syncthreads();
    float y = (x - mu_s) * rstd_s * g[c] + be[c];
    if (mode == 0) {
        outf[(size_t)node*H_ + c] = y;
        outh[(size_t)node*H_ + c] = (_Float16)y;
    } else if (mode == 1) {
        float yr = y + res[(size_t)node*H_ + c];
        outf[(size_t)node*H_ + c] = yr;
        outh[(size_t)node*H_ + c] = (_Float16)yr;
    } else {
        outh[(size_t)node*H_ + c] = (_Float16)y;
    }
}

// ---------------- head: WMMA GEMM1 + scalar 192->3 projection ----------------
__global__ __launch_bounds__(256)
void head_kernel(const _Float16* __restrict__ hh, const _Float16* __restrict__ w1T,
                 const float* __restrict__ b1, const float* __restrict__ w2,
                 const float* __restrict__ b2, float* __restrict__ delta, int Mtot) {
    __shared__ _Float16 T1[TILE_E * H_];
    const int wave = threadIdx.x >> 5, lane = threadIdx.x & 31;
    const int hf = lane >> 4, lrow = lane & 15;
    const int rb = wave >> 1, ch = wave & 1;
    const int row0 = blockIdx.x * TILE_E;

    v8f vz = {};
    v8f acc[6];
#pragma unroll
    for (int cb = 0; cb < 6; ++cb) acc[cb] = vz;

    int arow = row0 + rb*16 + lrow;
    bool rowok = (arow < Mtot);
    const _Float16* A = hh + (size_t)arow * H_;
    for (int kc = 0; kc < H_; kc += 32) {
        FragH a;
#pragma unroll
        for (int v = 0; v < 8; ++v)
            a.u[v] = rowok ? *(const unsigned int*)(A + kc + kmap_(v, hf)) : 0u;
#pragma unroll
        for (int cb = 0; cb < 6; ++cb) {
            int cbg = ch*6 + cb;
            FragH bf; load_frag(bf, w1T + (size_t)(cbg*16 + lrow)*H_ + kc, hf);
            acc[cb] = __builtin_amdgcn_wmma_f32_16x16x32_f16(
                false, a.h, false, bf.h, (short)0, acc[cb], false, false);
        }
    }
#pragma unroll
    for (int cb = 0; cb < 6; ++cb) {
        int col = (ch*6 + cb)*16 + lrow;
        float bb = b1[col];
#pragma unroll
        for (int r = 0; r < 8; ++r) {
            int m = r + 8*hf;
            T1[(rb*16 + m)*H_ + col] = (_Float16)gelu_(acc[cb][r] + bb);
        }
    }
    __syncthreads();
    for (int p = threadIdx.x; p < TILE_E * 3; p += blockDim.x) {
        int r = p / 3, c = p % 3;
        int rowg = row0 + r;
        if (rowg < Mtot) {
            float s = b2[c];
            for (int k = 0; k < H_; ++k)
                s += (float)T1[r*H_ + k] * w2[k*3 + c];
            delta[(size_t)rowg*3 + c] = s;
        }
    }
}

// ---------------- output ----------------
__global__ void outbase_kernel(const float* __restrict__ u, const float* __restrict__ am,
                               float* __restrict__ out) {
    int idx = blockIdx.x * blockDim.x + threadIdx.x;
    if (idx >= G_ * N_ * C_) return;
    int b = idx / (K_ * N_ * C_);
    int n = (idx / C_) % N_;
    out[idx] = u[idx] * (1.f - am[b*N_ + n]);
}

__global__ void scatter_kernel(const float* __restrict__ delta, const float* __restrict__ masksel,
                               const int* __restrict__ topidx, const float* __restrict__ am,
                               float* __restrict__ out) {
    int idx = blockIdx.x * blockDim.x + threadIdx.x;
    if (idx >= G_ * S_ * C_) return;
    int g = idx / (S_ * C_);
    int rem = idx % (S_ * C_);
    int s = rem / C_, c = rem % C_;
    int b = g / K_;
    int n = topidx[b*S_ + s];
    float v = delta[idx] * masksel[b*S_ + s] * (1.f - am[b*N_ + n]);
    out[((size_t)g*N_ + n)*C_ + c] += v;
}

// ---------------- host launch ----------------
extern "C" void kernel_launch(void* const* d_in, const int* in_sizes, int n_in,
                              void* d_out, int out_size, void* d_ws, size_t ws_size,
                              hipStream_t stream) {
    const float* u_base = (const float*)d_in[0];
    const float* pos    = (const float*)d_in[1];
    const float* vin    = (const float*)d_in[2];
    const float* am     = (const float*)d_in[3];
    const float* c0w1   = (const float*)d_in[4];
    const float* c0b1   = (const float*)d_in[5];
    const float* c0w2   = (const float*)d_in[6];
    const float* c0b2   = (const float*)d_in[7];
    const float* c0g    = (const float*)d_in[8];
    const float* c0be   = (const float*)d_in[9];
    const float* cw1    = (const float*)d_in[10];
    const float* cb1    = (const float*)d_in[11];
    const float* cw2    = (const float*)d_in[12];
    const float* cb2    = (const float*)d_in[13];
    const float* cg     = (const float*)d_in[14];
    const float* cbe    = (const float*)d_in[15];
    const float* hg     = (const float*)d_in[16];
    const float* hbe    = (const float*)d_in[17];
    const float* hw1    = (const float*)d_in[18];
    const float* hb1    = (const float*)d_in[19];
    const float* hw2    = (const float*)d_in[20];
    const float* hb2    = (const float*)d_in[21];
    float* out = (float*)d_out;

    char* ws = (char*)d_ws;
    size_t off = 0;
    auto carve = [&](size_t bytes) -> void* {
        void* p = ws + off;
        off += (bytes + 255) & ~(size_t)255;
        return p;
    };
    float* varpp   = (float*)carve((size_t)B_*N_*4);
    float* stats   = (float*)carve((size_t)B_*2*4);
    float* smask   = (float*)carve((size_t)B_*N_*4);
    int*   topidx  = (int*)  carve((size_t)B_*S_*4);
    float* possel  = (float*)carve((size_t)B_*S_*3*4);
    float* masksel = (float*)carve((size_t)B_*S_*4);
    float* x0      = (float*)carve((size_t)G_*S_*10*4);
    int*   knnW    = (int*)  carve((size_t)B_*S_*KNN_*4);
    float* eaW     = (float*)carve((size_t)B_*E_*4*4);
    _Float16* ea16 = (_Float16*)carve((size_t)B_*E_*32*2);
    float* hA      = (float*)carve((size_t)G_*S_*H_*4);
    float* hB      = (float*)carve((size_t)G_*S_*H_*4);
    float* agg     = (float*)carve((size_t)G_*S_*H_*4);
    _Float16* h16  = (_Float16*)carve((size_t)G_*S_*H_*2);
    _Float16* hh16 = (_Float16*)carve((size_t)G_*S_*H_*2);
    float* delta   = (float*)carve((size_t)G_*S_*3*4);
    _Float16* w1T0 = (_Float16*)carve((size_t)H_*KP0_*2);
    _Float16* w1Tl = (_Float16*)carve((size_t)9*H_*KPR_*2);
    _Float16* w2T  = (_Float16*)carve((size_t)10*H_*H_*2);
    _Float16* hw1T = (_Float16*)carve((size_t)H_*H_*2);

    const size_t aggBytes = (size_t)G_*S_*H_*4;

    // preprocessing
    varpp_kernel<<<(B_*N_ + 255)/256, 256, 0, stream>>>(vin, varpp);
    stats_kernel<<<B_, 256, 0, stream>>>(varpp, stats);
    softmask_kernel<<<(B_*N_ + 255)/256, 256, 0, stream>>>(varpp, stats, am, smask);
    topk_sort_kernel<<<B_, 1024, 0, stream>>>(varpp, topidx);
    gather_kernel<<<(B_*S_ + 255)/256, 256, 0, stream>>>(u_base, pos, vin, varpp, smask,
                                                         topidx, possel, masksel, x0);
    {
        dim3 kgrid((S_ + 255)/256, B_);
        knn_kernel<<<kgrid, 256, 0, stream>>>(possel, knnW, eaW, ea16);
    }

    // weight prep: W1 folded (hi coeff = W1a - W1b), transpose + pad + f16
    prep_w1fold_kernel<<<256, 256, 0, stream>>>(c0w1, w1T0, 10, KP0_, 1, F0_*H_, H_*KP0_);
    prep_w1fold_kernel<<<256, 256, 0, stream>>>(cw1, w1Tl, H_, KPR_, 9, FR_*H_, H_*KPR_);
    prep_w_kernel<<<256, 256, 0, stream>>>(c0w2, w2T, H_, H_, 1, H_*H_, H_*H_);
    prep_w_kernel<<<256, 256, 0, stream>>>(cw2, w2T + (size_t)H_*H_, H_, H_, 9, H_*H_, H_*H_);
    prep_w_kernel<<<256, 256, 0, stream>>>(hw1, hw1T, H_, H_, 1, H_*H_, H_*H_);

    dim3 cgrid((E_ + TILE_E - 1)/TILE_E, G_);
    size_t lds0 = (size_t)TILE_E * H_   * 2 + TILE_E * 4;  // maxKp = 192 (layer 0)
    size_t ldsR = (size_t)TILE_E * KPR_ * 2 + TILE_E * 4;  // 416

    // layer 0
    hipMemsetAsync(agg, 0, aggBytes, stream);
    conv_edge_mlp<<<cgrid, 256, lds0, stream>>>(x0, h16, 10, knnW, eaW, ea16,
                                                w1T0, KP0_, c0b1, w2T, c0b2, agg);
    ln_kernel<<<G_*S_, H_, 0, stream>>>(agg, (const float*)nullptr, c0g, c0be, hA, h16, 0);
    // layers 1..9 (residual); gather reads f16 mirror h16 via async DMA
    float* hcur = hA; float* hnext = hB;
    for (int l = 0; l < 9; ++l) {
        hipMemsetAsync(agg, 0, aggBytes, stream);
        conv_edge_mlp<<<cgrid, 256, ldsR, stream>>>(hcur, h16, H_, knnW, eaW, ea16,
            w1Tl + (size_t)l*H_*KPR_, KPR_, cb1 + l*H_,
            w2T + (size_t)(l+1)*H_*H_, cb2 + l*H_, agg);
        ln_kernel<<<G_*S_, H_, 0, stream>>>(agg, hcur, cg + l*H_, cbe + l*H_, hnext, h16, 1);
        float* t = hcur; hcur = hnext; hnext = t;
    }
    // head
    ln_kernel<<<G_*S_, H_, 0, stream>>>(hcur, (const float*)nullptr, hg, hbe,
                                        (float*)nullptr, hh16, 2);
    head_kernel<<<(G_*S_ + TILE_E - 1)/TILE_E, 256, 0, stream>>>(hh16, hw1T, hb1, hw2, hb2,
                                                                 delta, G_*S_);
    // output assembly
    outbase_kernel<<<(G_*N_*C_ + 255)/256, 256, 0, stream>>>(u_base, am, out);
    scatter_kernel<<<(G_*S_*C_ + 255)/256, 256, 0, stream>>>(delta, masksel, topidx, am, out);
}
